// GATModel_25434796327171
// MI455X (gfx1250) — compile-verified, matrix operand order checked by reading, not run on hardware
//
#include <hip/hip_runtime.h>
#include <hip/hip_bf16.h>
#include <math.h>

// Problem constants (from the reference)
#define N_NODES 50000
#define N_EDGES 800000
#define F_IN    128
#define CC      64
#define EDIM    16
#define NG      64
#define NOUT    8

typedef __attribute__((ext_vector_type(16))) __bf16 v16bf;
typedef __attribute__((ext_vector_type(8)))  float  v8f;

union ABu {
    v16bf v;
    uint4 q[2];
};

__device__ __forceinline__ unsigned short f2bf(float f) {
    unsigned u = __float_as_uint(f);
    u += 0x7fffu + ((u >> 16) & 1u);   // round-to-nearest-even
    return (unsigned short)(u >> 16);
}

// ---------------- conversion / fill ----------------

__global__ void k_f32_to_bf16(const float* __restrict__ src,
                              unsigned short* __restrict__ dst, int n) {
    int i = blockIdx.x * blockDim.x + threadIdx.x;
    if (i < n) dst[i] = f2bf(src[i]);
}

__global__ void k_fill(float* __restrict__ p, float v, int n) {
    int i = blockIdx.x * blockDim.x + threadIdx.x;
    if (i < n) p[i] = v;
}

// ---------------- WMMA GEMM: D[M,Nc] = A[M,K](bf16) @ B[Nc,K](bf16)^T (+bias) ----------------
// One wave per 16x16 output tile. K multiple of 32, M multiple of 16.
// A-operand layout (ISA 7.12.2, 16-bit A 16x32): lane m = l&15, khalf = (l>>4)*8,
//   elems 0..7 -> K = khalf..khalf+7, elems 8..15 -> K = 16+khalf .. 16+khalf+7.
// B loaded symmetrically with lane = output column (B row of weights, column-major K).
// C/D (f32 16x16): lane n = l&15, elem j -> row m = j + 8*(l>>4).

__global__ void k_gemm_bf16_wmma(const unsigned short* __restrict__ A,
                                 const unsigned short* __restrict__ B,
                                 const float* __restrict__ bias,
                                 float* __restrict__ D,
                                 int M, int Nc, int K) {
    int lane  = threadIdx.x & 31;
    int wave  = threadIdx.x >> 5;
    int mtile = blockIdx.x * (blockDim.x >> 5) + wave;
    int ntile = blockIdx.y;
    if (mtile * 16 >= M) return;            // wave-uniform: EXEC stays all-ones

    int mrow  = mtile * 16 + (lane & 15);   // A row this lane feeds
    int ncol  = ntile * 16 + (lane & 15);   // weight row == output column
    int khalf = (lane >> 4) * 8;

    const unsigned short* arow = A + (size_t)mrow * K;
    const unsigned short* brow = B + (size_t)ncol * K;

    v8f acc = {};
    for (int k0 = 0; k0 < K; k0 += 32) {
        ABu a, b;
        a.q[0] = *(const uint4*)(arow + k0 + khalf);
        a.q[1] = *(const uint4*)(arow + k0 + 16 + khalf);
        b.q[0] = *(const uint4*)(brow + k0 + khalf);
        b.q[1] = *(const uint4*)(brow + k0 + 16 + khalf);
        acc = __builtin_amdgcn_wmma_f32_16x16x32_bf16(
            /*neg_a=*/false, a.v, /*neg_b=*/false, b.v,
            /*c_mod=*/(short)0, acc, /*reuse_a=*/false, /*reuse_b=*/false);
    }

    float bv = bias ? bias[ncol] : 0.0f;
    int baseRow = mtile * 16 + ((lane >> 4) ? 8 : 0);
#pragma unroll
    for (int j = 0; j < 8; ++j)
        D[(size_t)(baseRow + j) * Nc + ncol] = acc[j] + bv;
}

// ---------------- attention pieces ----------------

// a_src[n,h] = <h[n,h,:], att_src[h,:]>, same for dst
__global__ void k_att_nodes(const float* __restrict__ h,
                            const float* __restrict__ att_src,
                            const float* __restrict__ att_dst,
                            float* __restrict__ a_src, float* __restrict__ a_dst,
                            int H) {
    int idx = blockIdx.x * blockDim.x + threadIdx.x;
    if (idx >= N_NODES * H) return;
    int node = idx / H, hd = idx - node * H;
    const float* hp = h + (size_t)node * H * CC + hd * CC;
    const float* as = att_src + hd * CC;
    const float* ad = att_dst + hd * CC;
    float s = 0.f, d = 0.f;
#pragma unroll 8
    for (int c = 0; c < CC; ++c) { float v = hp[c]; s += v * as[c]; d += v * ad[c]; }
    a_src[idx] = s;
    a_dst[idx] = d;
}

// M[f,h] = sum_c W_edge[h*C+c, f] * att_edge[h*C+c]   (tiny: EDIM*H <= 64)
__global__ void k_edge_proj(const float* __restrict__ W_edge,
                            const float* __restrict__ att_edge,
                            float* __restrict__ Mo, int H) {
    int t = threadIdx.x;
    if (t >= EDIM * H) return;
    int f = t % EDIM, h = t / EDIM;
    float s = 0.f;
    for (int c = 0; c < CC; ++c)
        s += W_edge[(size_t)(h * CC + c) * EDIM + f] * att_edge[h * CC + c];
    Mo[f * H + h] = s;
}

// alpha = lrelu(a_src[src]+a_dst[dst]+e_attr@M); store; atomic-max into amax[dst]
__global__ void k_edge_alpha(const int* __restrict__ src, const int* __restrict__ dst,
                             const float* __restrict__ eattr,
                             const float* __restrict__ a_src, const float* __restrict__ a_dst,
                             const float* __restrict__ Mo,
                             float* __restrict__ exb, float* __restrict__ amax, int H) {
    int idx = blockIdx.x * blockDim.x + threadIdx.x;
    if (idx >= N_EDGES * H) return;
    int e = idx / H, h = idx - e * H;
    int s = src[e], d = dst[e];
    float al = a_src[s * H + h] + a_dst[d * H + h];
    const float* ea = eattr + (size_t)e * EDIM;
#pragma unroll
    for (int f = 0; f < EDIM; ++f) al += ea[f] * Mo[f * H + h];
    al = (al > 0.f) ? al : 0.2f * al;           // leaky_relu(0.2)
    exb[idx] = al;
    // float atomic max via sign-magnitude trick (amax initialized to -INF)
    float* addr = amax + d * H + h;
    if (al >= 0.f) atomicMax((int*)addr, __float_as_int(al));
    else           atomicMin((unsigned int*)addr, __float_as_uint(al));
}

// ex = exp(alpha - amax[dst]); denom[dst] += ex
__global__ void k_edge_exp(const int* __restrict__ dst,
                           float* __restrict__ exb,
                           const float* __restrict__ amax,
                           float* __restrict__ denom, int H) {
    int idx = blockIdx.x * blockDim.x + threadIdx.x;
    if (idx >= N_EDGES * H) return;
    int e = idx / H, h = idx - e * H;
    float ex = __expf(exb[idx] - amax[dst[e] * H + h]);
    exb[idx] = ex;
    atomicAdd(denom + dst[e] * H + h, ex);
}

// out[dst, j] += h[src, j] * (ex / denom[dst])
__global__ void k_aggregate(const int* __restrict__ src, const int* __restrict__ dst,
                            const float* __restrict__ hbuf,
                            const float* __restrict__ exb, const float* __restrict__ denom,
                            float* __restrict__ out, int H, int hcShift, int total) {
    int idx = blockIdx.x * blockDim.x + threadIdx.x;
    if (idx >= total) return;
    int HC = 1 << hcShift;
    int e = idx >> hcShift;
    int j = idx & (HC - 1);
    int h = j >> 6;                 // C == 64
    int s = src[e], d = dst[e];
    float attn = exb[e * H + h] / (denom[d * H + h] + 1e-16f);
    atomicAdd(out + (size_t)d * HC + j, hbuf[(size_t)s * HC + j] * attn);
}

// x = bn(relu(out + bias)) + res ; also emit bf16(x) for next layer's WMMA
__global__ void k_post(float* __restrict__ xio,
                       const float* __restrict__ resb,
                       const float* __restrict__ bias,
                       const float* __restrict__ gamma, const float* __restrict__ beta,
                       const float* __restrict__ mean, const float* __restrict__ var,
                       unsigned short* __restrict__ Xb,
                       int doutShift, int total) {
    int idx = blockIdx.x * blockDim.x + threadIdx.x;
    if (idx >= total) return;
    int dmask = (1 << doutShift) - 1;
    int j = idx & dmask;
    float v = xio[idx] + bias[j];
    v = fmaxf(v, 0.f);
    v = (v - mean[j]) * rsqrtf(var[j] + 1e-5f) * gamma[j] + beta[j];
    v += resb[idx];
    xio[idx] = v;
    Xb[idx] = f2bf(v);
}

// ---------------- pooling + MLP head ----------------

__global__ void k_count(const int* __restrict__ batch, float* __restrict__ cnt) {
    int n = blockIdx.x * blockDim.x + threadIdx.x;
    if (n < N_NODES) atomicAdd(cnt + batch[n], 1.0f);
}

__global__ void k_pool(const int* __restrict__ batch, const float* __restrict__ x,
                       float* __restrict__ pool) {
    int idx = blockIdx.x * blockDim.x + threadIdx.x;
    if (idx >= N_NODES * CC) return;
    int n = idx >> 6, j = idx & 63;
    atomicAdd(pool + batch[n] * CC + j, x[idx]);
}

__global__ void k_head(const float* __restrict__ pool, const float* __restrict__ cnt,
                       const float* __restrict__ fc1w, const float* __restrict__ fc1b,
                       const float* __restrict__ fc2w, const float* __restrict__ fc2b,
                       float* __restrict__ out) {
    int g = threadIdx.x;
    if (g >= NG) return;
    float inv = 1.0f / fmaxf(cnt[g], 1.0f);
    float gf[CC];
    for (int c = 0; c < CC; ++c) gf[c] = pool[g * CC + c] * inv;
    float h1[CC / 2];
    for (int k = 0; k < CC / 2; ++k) {
        float s = fc1b[k];
        for (int c = 0; c < CC; ++c) s += fc1w[k * CC + c] * gf[c];
        h1[k] = fmaxf(s, 0.f);
    }
    for (int o = 0; o < NOUT; ++o) {
        float s = fc2b[o];
        for (int k = 0; k < CC / 2; ++k) s += fc2w[o * (CC / 2) + k] * h1[k];
        out[g * NOUT + o] = s;
    }
}

// ---------------- host orchestration ----------------

#define CDIV(a, b) (((a) + (b) - 1) / (b))

extern "C" void kernel_launch(void* const* d_in, const int* in_sizes, int n_in,
                              void* d_out, int out_size, void* d_ws, size_t ws_size,
                              hipStream_t stream) {
    (void)in_sizes; (void)n_in; (void)out_size; (void)ws_size;

    const float* x_in  = (const float*)d_in[0];
    const int*   eidx  = (const int*)d_in[1];
    const int*   srcI  = eidx;            // edge_index[0]
    const int*   dstI  = eidx + N_EDGES;  // edge_index[1]
    const float* eattr = (const float*)d_in[2];
    const int*   batch = (const int*)d_in[3];
    float*       outp  = (float*)d_out;

    // workspace carve-out
    char* ws = (char*)d_ws;
    size_t off = 0;
    auto carve = [&](size_t bytes) -> char* {
        char* p = ws + off;
        off = (off + bytes + 255) & ~(size_t)255;
        return p;
    };
    unsigned short* Xb   = (unsigned short*)carve((size_t)N_NODES * 256 * 2);
    float*          hbuf = (float*)carve((size_t)N_NODES * 256 * 4);
    float*          resb = (float*)carve((size_t)N_NODES * 256 * 4);
    float*          xbuf = (float*)carve((size_t)N_NODES * 256 * 4);
    unsigned short* Wb   = (unsigned short*)carve((size_t)256 * 256 * 2);
    unsigned short* Wr   = (unsigned short*)carve((size_t)256 * 256 * 2);
    float*          aS   = (float*)carve((size_t)N_NODES * 4 * 4);
    float*          aD   = (float*)carve((size_t)N_NODES * 4 * 4);
    float*          amax = (float*)carve((size_t)N_NODES * 4 * 4);
    float*          den  = (float*)carve((size_t)N_NODES * 4 * 4);
    float*          exb  = (float*)carve((size_t)N_EDGES * 4 * 4);
    float*          Mbuf = (float*)carve(256);
    float*          pool = (float*)carve((size_t)NG * CC * 4);
    float*          cnt  = (float*)carve((size_t)NG * 4);

    const int dins[3] = {F_IN, 4 * CC, 4 * CC};
    const int Hs[3]   = {4, 4, 1};

    const int MT = N_NODES / 16;  // 3125 exact row tiles

    for (int L = 0; L < 3; ++L) {
        int din = dins[L], H = Hs[L];
        int HC = H * CC;                       // gat width == dout (concat / H==1 mean)
        int dout = HC;
        int hcShift = (HC == 256) ? 8 : 6;

        const float* W        = (const float*)d_in[4 + L * 12 + 0];
        const float* att_src  = (const float*)d_in[4 + L * 12 + 1];
        const float* att_dst  = (const float*)d_in[4 + L * 12 + 2];
        const float* W_edge   = (const float*)d_in[4 + L * 12 + 3];
        const float* att_edge = (const float*)d_in[4 + L * 12 + 4];
        const float* bias     = (const float*)d_in[4 + L * 12 + 5];
        const float* res_w    = (const float*)d_in[4 + L * 12 + 6];
        const float* res_b    = (const float*)d_in[4 + L * 12 + 7];
        const float* gamma    = (const float*)d_in[4 + L * 12 + 8];
        const float* beta     = (const float*)d_in[4 + L * 12 + 9];
        const float* mean     = (const float*)d_in[4 + L * 12 + 10];
        const float* var      = (const float*)d_in[4 + L * 12 + 11];

        if (L == 0) {
            int n = N_NODES * F_IN;
            k_f32_to_bf16<<<CDIV(n, 256), 256, 0, stream>>>(x_in, Xb, n);
        }

        // h = x @ W^T   [N, HC]
        {
            int n = HC * din;
            k_f32_to_bf16<<<CDIV(n, 256), 256, 0, stream>>>(W, Wb, n);
            dim3 grid(CDIV(MT, 4), HC / 16);
            k_gemm_bf16_wmma<<<grid, 128, 0, stream>>>(Xb, Wb, nullptr, hbuf,
                                                       N_NODES, HC, din);
        }
        // res = x @ res_w^T + res_b   [N, dout]
        {
            int n = dout * din;
            k_f32_to_bf16<<<CDIV(n, 256), 256, 0, stream>>>(res_w, Wr, n);
            dim3 grid(CDIV(MT, 4), dout / 16);
            k_gemm_bf16_wmma<<<grid, 128, 0, stream>>>(Xb, Wr, res_b, resb,
                                                       N_NODES, dout, din);
        }

        k_edge_proj<<<1, 64, 0, stream>>>(W_edge, att_edge, Mbuf, H);

        {
            int n = N_NODES * H;
            k_att_nodes<<<CDIV(n, 256), 256, 0, stream>>>(hbuf, att_src, att_dst, aS, aD, H);
            k_fill<<<CDIV(n, 256), 256, 0, stream>>>(amax, -INFINITY, n);
            k_fill<<<CDIV(n, 256), 256, 0, stream>>>(den, 0.0f, n);
        }
        {
            int n = N_NODES * HC;
            k_fill<<<CDIV(n, 256), 256, 0, stream>>>(xbuf, 0.0f, n);
        }

        {
            int n = N_EDGES * H;
            k_edge_alpha<<<CDIV(n, 256), 256, 0, stream>>>(srcI, dstI, eattr, aS, aD,
                                                           Mbuf, exb, amax, H);
            k_edge_exp<<<CDIV(n, 256), 256, 0, stream>>>(dstI, exb, amax, den, H);
        }
        {
            int total = N_EDGES * HC;   // <= 204.8M, fits int
            k_aggregate<<<CDIV(total, 256), 256, 0, stream>>>(srcI, dstI, hbuf, exb, den,
                                                              xbuf, H, hcShift, total);
        }
        {
            int total = N_NODES * dout;
            k_post<<<CDIV(total, 256), 256, 0, stream>>>(xbuf, resb, bias, gamma, beta,
                                                         mean, var, Xb, hcShift, total);
        }
    }

    // global mean pool + MLP head
    const float* fc1w = (const float*)d_in[40];
    const float* fc1b = (const float*)d_in[41];
    const float* fc2w = (const float*)d_in[42];
    const float* fc2b = (const float*)d_in[43];

    k_fill<<<CDIV(NG * CC, 256), 256, 0, stream>>>(pool, 0.0f, NG * CC);
    k_fill<<<1, 64, 0, stream>>>(cnt, 0.0f, NG);
    k_count<<<CDIV(N_NODES, 256), 256, 0, stream>>>(batch, cnt);
    k_pool<<<CDIV(N_NODES * CC, 256), 256, 0, stream>>>(batch, xbuf, pool);
    k_head<<<1, 64, 0, stream>>>(pool, cnt, fc1w, fc1b, fc2w, fc2b, outp);
}
